// BiLSTM_87402584474199
// MI455X (gfx1250) — compile-verified, matrix operand order checked by reading, not run on hardware
//
#include <hip/hip_runtime.h>
#include <hip/hip_bf16.h>

#define N 512
#define DIN 300
#define H 256
#define G4 1024   // 4*H
#define L 50
#define LP 64     // padded label dim (multiple of 16)
#define TI 16
#define TJ 8
#define MT 128    // TI*TJ pair-rows per workgroup

// LDS layout for k_pair_mlp (dynamic shared, no static LDS in that kernel)
#define LDS_H1   0          // [MT][H]  f16  = 64 KB (reused for h2, then lbuf)
#define LDS_W2   65536      // [H][H]   f16  = 128 KB (TDM-staged)
#define LDS_W3   196608     // [LP][H]  f16  = 32 KB  (TDM-staged)
#define LDS_TOT  229376

typedef __attribute__((ext_vector_type(16))) _Float16 v16h;
typedef __attribute__((ext_vector_type(8)))  float    v8f;
typedef __attribute__((ext_vector_type(4)))  unsigned int u32x4;
typedef __attribute__((ext_vector_type(8)))  int      i32x8;
typedef __attribute__((ext_vector_type(4)))  int      i32x4;

union FragH { v16h h; uint4 u[2]; };

__device__ __forceinline__ float sigmoidf_(float v) { return 1.0f / (1.0f + __expf(-v)); }

// ---------------------------------------------------------------------------
// TDM: async 2D tile load (16-bit elements) global -> LDS via D# descriptor.
// Descriptor layout per CDNA5 ISA.
// ---------------------------------------------------------------------------
__device__ __forceinline__ void tdm_load_2d_f16(const void* gsrc, unsigned lds_off,
                                                unsigned rows, unsigned cols,
                                                unsigned stride_elems) {
  unsigned long long ga = (unsigned long long)gsrc;
  u32x4 g0 = { 1u,                                   // count=1 (valid), no gather
               lds_off,                              // lds_addr [63:32]
               (unsigned)ga,                         // global_addr lo
               (unsigned)((ga >> 32) & 0x1FFFFFFu)   // global_addr hi (57-bit)
                   | 0x80000000u };                  // type=2 ("image") [127:126]
  i32x8 g1;
  g1[0] = (int)(1u << 16);                 // data_size=1 -> 2 bytes; wg_mask=0
  g1[1] = (int)((cols & 0xFFFFu) << 16);   // [63:48] tensor_dim0 lo16
  g1[2] = (int)((rows & 0xFFFFu) << 16);   // [79:64] dim0 hi16=0 | [95:80] tensor_dim1 lo16
  g1[3] = (int)((cols & 0xFFFFu) << 16);   // [111:96] dim1 hi16=0 | [127:112] tile_dim0
  g1[4] = (int)(rows & 0xFFFFu);           // [143:128] tile_dim1 | tile_dim2=0
  g1[5] = (int)stride_elems;               // tensor_dim0_stride lo32
  g1[6] = 0;                               // stride0 hi16 | stride1 lo16
  g1[7] = 0;                               // stride1 hi32
  i32x4 gz = { 0, 0, 0, 0 };
#if defined(__clang_major__) && __clang_major__ >= 23
  i32x8 gz8 = { 0, 0, 0, 0, 0, 0, 0, 0 };
  __builtin_amdgcn_tensor_load_to_lds(g0, g1, gz, gz, gz8, 0);
#else
  __builtin_amdgcn_tensor_load_to_lds(g0, g1, gz, gz, 0);
#endif
}

// ---------------------------------------------------------------------------
// K1: xb[dir][t][c] = x[t] . Wih[dir][c] + bih[c] + bhh[c]
// ---------------------------------------------------------------------------
__global__ void k_input_proj(const float* __restrict__ x,
                             const float* __restrict__ Wih_f, const float* __restrict__ bih_f,
                             const float* __restrict__ bhh_f,
                             const float* __restrict__ Wih_b, const float* __restrict__ bih_b,
                             const float* __restrict__ bhh_b,
                             float* __restrict__ xb) {
  int t = blockIdx.x * blockDim.x + threadIdx.x;
  if (t >= 2 * N * G4) return;
  int dir = t / (N * G4);
  int rem = t - dir * (N * G4);
  int row = rem / G4;
  int col = rem - row * G4;
  const float* Wih = dir ? Wih_b : Wih_f;
  const float* bih = dir ? bih_b : bih_f;
  const float* bhh = dir ? bhh_b : bhh_f;
  const float* xr = x + row * DIN;
  const float* wr = Wih + col * DIN;
  float acc = bih[col] + bhh[col];
#pragma unroll 4
  for (int k = 0; k < DIN; ++k) acc += xr[k] * wr[k];
  xb[t] = acc;
}

// ---------------------------------------------------------------------------
// K2: sequential LSTM, one block per direction, h kept in LDS, c in registers
// ---------------------------------------------------------------------------
__global__ void __launch_bounds__(1024) k_lstm(const float* __restrict__ xb,
                                               const float* __restrict__ Whh_f,
                                               const float* __restrict__ Whh_b,
                                               float* __restrict__ out /* [N][2H] */) {
  __shared__ float h_s[H];
  __shared__ float g_s[G4];
  const int dir = blockIdx.x;
  const int tid = threadIdx.x;
  const float* Whh = dir ? Whh_b : Whh_f;
  const float* xbd = xb + (size_t)dir * N * G4;
  float c = 0.f;
  if (tid < H) h_s[tid] = 0.f;
  __syncthreads();
  const float4* wrow = (const float4*)(Whh + tid * H);
  for (int s = 0; s < N; ++s) {
    int tstep = dir ? (N - 1 - s) : s;
    float acc = xbd[tstep * G4 + tid];
#pragma unroll 8
    for (int k4 = 0; k4 < H / 4; ++k4) {
      float4 w = wrow[k4];
      const float* hh = &h_s[k4 * 4];
      acc += w.x * hh[0] + w.y * hh[1] + w.z * hh[2] + w.w * hh[3];
    }
    g_s[tid] = acc;
    __syncthreads();
    if (tid < H) {
      float ig = sigmoidf_(g_s[tid]);
      float fg = sigmoidf_(g_s[H + tid]);
      float gg = tanhf(g_s[2 * H + tid]);
      float og = sigmoidf_(g_s[3 * H + tid]);
      c = fg * c + ig * gg;
      float hn = og * tanhf(c);
      h_s[tid] = hn;
      out[tstep * (2 * H) + dir * H + tid] = hn;
    }
    __syncthreads();
  }
}

// ---------------------------------------------------------------------------
// K3: a = out @ W1[:, :2H].T ; b = out @ W1[:, 2H:].T
// ---------------------------------------------------------------------------
__global__ void k_proj_ab(const float* __restrict__ hout, const float* __restrict__ W1,
                          float* __restrict__ amat, float* __restrict__ bmat) {
  int t = blockIdx.x * blockDim.x + threadIdx.x;
  if (t >= 2 * N * H) return;
  int which = t / (N * H);
  int rem = t - which * (N * H);
  int row = rem / H;
  int hc = rem - row * H;
  const float* o = hout + row * (2 * H);
  const float* w = W1 + hc * (4 * H) + which * (2 * H);
  float acc = 0.f;
#pragma unroll 4
  for (int k = 0; k < 2 * H; ++k) acc += o[k] * w[k];
  (which ? bmat : amat)[row * H + hc] = acc;
}

// ---------------------------------------------------------------------------
// K4: convert W2 -> f16, W3 -> f16 zero-padded to [LP][H]
// ---------------------------------------------------------------------------
__global__ void k_convert(const float* __restrict__ W2, const float* __restrict__ W3,
                          _Float16* __restrict__ W2h, _Float16* __restrict__ W3h) {
  int t = blockIdx.x * blockDim.x + threadIdx.x;
  if (t < H * H) W2h[t] = (_Float16)W2[t];
  int t3 = t - H * H;
  if (t3 >= 0 && t3 < LP * H)
    W3h[t3] = (t3 < L * H) ? (_Float16)W3[t3] : (_Float16)0.f;
}

// ---------------------------------------------------------------------------
// K5: fused pairwise MLP + log_softmax. 8 waves, 224KB dynamic LDS, WMMA f16,
//     weights staged into LDS by the Tensor Data Mover (async, TENSORcnt).
// ---------------------------------------------------------------------------
__global__ void __launch_bounds__(256) k_pair_mlp(
    const float* __restrict__ amat, const float* __restrict__ bmat,
    const float* __restrict__ b1, const float* __restrict__ b2,
    const float* __restrict__ b3,
    const _Float16* __restrict__ W2h, const _Float16* __restrict__ W3h,
    float* __restrict__ outp /* [N*N][L] */) {
  extern __shared__ __align__(16) char smem[];
  _Float16* h1  = (_Float16*)(smem + LDS_H1);  // [MT][H] f16 (reused for h2)
  float*    lbuf = (float*)(smem + LDS_H1);    // [MT][LP] f32 (after h2 consumed)
  const _Float16* W2s = (const _Float16*)(smem + LDS_W2);  // [H][H] f16
  const _Float16* W3s = (const _Float16*)(smem + LDS_W3);  // [LP][H] f16

  const int tid = threadIdx.x;
  const int jb = blockIdx.x;  // N/TJ = 64
  const int ib = blockIdx.y;  // N/TI = 32
  const int wave = tid >> 5;
  const int lane = tid & 31;
  const int lr = lane & 15;
  const int lg = lane >> 4;

  // Kick off TDM weight staging from one wave; overlaps with phase 1 VALU work.
  if (wave == 0) {
    tdm_load_2d_f16(W2h, LDS_W2, H, H, H);
    tdm_load_2d_f16(W3h, LDS_W3, LP, H, H);
  }

  // Phase 1: h1[m,k] = relu(a[i(m),k] + b[j(m),k] + b1[k]) in f16 LDS
  for (int idx = tid; idx < MT * H; idx += 256) {
    int m = idx >> 8;
    int k = idx & (H - 1);
    int il = m >> 3;
    int jl = m & (TJ - 1);
    float v = amat[(ib * TI + il) * H + k] + bmat[(jb * TJ + jl) * H + k] + b1[k];
    h1[idx] = (_Float16)fmaxf(v, 0.f);
  }
  if (wave == 0) __builtin_amdgcn_s_wait_tensorcnt(0);  // TDM transfers done
  __syncthreads();

  // Phase 2: h2 = relu(h1 @ W2^T + b2)   (128x256x256 WMMA GEMM, A+B in LDS)
  {
    const int mg = wave & 1;   // 2 m-halves of 4 tiles
    const int ng = wave >> 1;  // 4 n-quarters of 4 tiles
    v8f acc[4][4] = {};
    for (int kt = 0; kt < 8; ++kt) {
      const int kb = kt * 32 + lg * 8;  // 16-bit A-frag K striping
      FragH afr[4], bfr[4];
#pragma unroll
      for (int i = 0; i < 4; ++i) {
        const _Float16* p = h1 + ((mg * 4 + i) * 16 + lr) * H + kb;
        afr[i].u[0] = *(const uint4*)(p);
        afr[i].u[1] = *(const uint4*)(p + 16);
      }
#pragma unroll
      for (int i = 0; i < 4; ++i) {
        const _Float16* p = W2s + ((ng * 4 + i) * 16 + lr) * H + kb;
        bfr[i].u[0] = *(const uint4*)(p);
        bfr[i].u[1] = *(const uint4*)(p + 16);
      }
#pragma unroll
      for (int mi = 0; mi < 4; ++mi)
#pragma unroll
        for (int ni = 0; ni < 4; ++ni)
          acc[mi][ni] = __builtin_amdgcn_wmma_f32_16x16x32_f16(
              false, afr[mi].h, false, bfr[ni].h, (short)0, acc[mi][ni], false, false);
    }
    __syncthreads();  // everyone done reading h1
#pragma unroll
    for (int ni = 0; ni < 4; ++ni) {
      int ncol = (ng * 4 + ni) * 16 + lr;
      float bias = b2[ncol];
#pragma unroll
      for (int mi = 0; mi < 4; ++mi) {
        int mbase = (mg * 4 + mi) * 16 + lg * 8;
#pragma unroll
        for (int v = 0; v < 8; ++v) {
          float val = fmaxf(acc[mi][ni][v] + bias, 0.f);
          h1[(mbase + v) * H + ncol] = (_Float16)val;  // h2 in-place
        }
      }
    }
  }
  __syncthreads();

  // Phase 3: logits = h2 @ W3^T  (128x64x256 WMMA; wave w owns m-tile w)
  {
    v8f acc[4] = {};
    for (int kt = 0; kt < 8; ++kt) {
      const int kb = kt * 32 + lg * 8;
      FragH afr, bfr[4];
      {
        const _Float16* p = h1 + (wave * 16 + lr) * H + kb;
        afr.u[0] = *(const uint4*)(p);
        afr.u[1] = *(const uint4*)(p + 16);
      }
#pragma unroll
      for (int ni = 0; ni < 4; ++ni) {
        const _Float16* p = W3s + (ni * 16 + lr) * H + kb;
        bfr[ni].u[0] = *(const uint4*)(p);
        bfr[ni].u[1] = *(const uint4*)(p + 16);
      }
#pragma unroll
      for (int ni = 0; ni < 4; ++ni)
        acc[ni] = __builtin_amdgcn_wmma_f32_16x16x32_f16(
            false, afr.h, false, bfr[ni].h, (short)0, acc[ni], false, false);
    }
    __syncthreads();  // everyone done reading h2 before lbuf overwrite
#pragma unroll
    for (int ni = 0; ni < 4; ++ni) {
      int ncol = ni * 16 + lr;
#pragma unroll
      for (int v = 0; v < 8; ++v)
        lbuf[(wave * 16 + lg * 8 + v) * LP + ncol] = acc[ni][v];
    }
  }
  __syncthreads();

  // Phase 4: log_softmax over L=50 valid columns, write to d_out
  if (tid < MT) {
    const float* row = lbuf + tid * LP;
    float mx = -3.4e38f;
    for (int l = 0; l < L; ++l) mx = fmaxf(mx, row[l] + b3[l]);
    float s = 0.f;
    for (int l = 0; l < L; ++l) s += __expf(row[l] + b3[l] - mx);
    float lg2 = __logf(s);
    int il = tid >> 3, jl = tid & 7;
    size_t p = (size_t)(ib * TI + il) * N + (jb * TJ + jl);
    float* op = outp + p * L;
    for (int l = 0; l < L; ++l) op[l] = row[l] + b3[l] - mx - lg2;
  }
}

// ---------------------------------------------------------------------------
extern "C" void kernel_launch(void* const* d_in, const int* in_sizes, int n_in,
                              void* d_out, int out_size, void* d_ws, size_t ws_size,
                              hipStream_t stream) {
  (void)in_sizes; (void)n_in; (void)out_size; (void)ws_size;
  const float* x     = (const float*)d_in[0];
  const float* Wih_f = (const float*)d_in[1];
  const float* Whh_f = (const float*)d_in[2];
  const float* bih_f = (const float*)d_in[3];
  const float* bhh_f = (const float*)d_in[4];
  const float* Wih_b = (const float*)d_in[5];
  const float* Whh_b = (const float*)d_in[6];
  const float* bih_b = (const float*)d_in[7];
  const float* bhh_b = (const float*)d_in[8];
  const float* W1    = (const float*)d_in[9];
  const float* b1    = (const float*)d_in[10];
  const float* W2    = (const float*)d_in[11];
  const float* b2    = (const float*)d_in[12];
  const float* W3    = (const float*)d_in[13];
  const float* b3    = (const float*)d_in[14];

  char* ws = (char*)d_ws;
  size_t off = 0;
  float* xb   = (float*)(ws + off); off += (size_t)2 * N * G4 * 4;   // 4 MB
  float* hout = (float*)(ws + off); off += (size_t)N * 2 * H * 4;    // 1 MB
  float* amat = (float*)(ws + off); off += (size_t)N * H * 4;        // 0.5 MB
  float* bmat = (float*)(ws + off); off += (size_t)N * H * 4;        // 0.5 MB
  _Float16* W2h = (_Float16*)(ws + off); off += (size_t)H * H * 2;   // 128 KB
  _Float16* W3h = (_Float16*)(ws + off); off += (size_t)LP * H * 2;  // 32 KB

  k_input_proj<<<(2 * N * G4 + 255) / 256, 256, 0, stream>>>(
      x, Wih_f, bih_f, bhh_f, Wih_b, bih_b, bhh_b, xb);
  k_lstm<<<2, 1024, 0, stream>>>(xb, Whh_f, Whh_b, hout);
  k_proj_ab<<<(2 * N * H + 255) / 256, 256, 0, stream>>>(hout, W1, amat, bmat);
  k_convert<<<(H * H + LP * H + 255) / 256, 256, 0, stream>>>(W2, W3, W2h, W3h);
  k_pair_mlp<<<dim3(N / TJ, N / TI), 256, LDS_TOT, stream>>>(
      amat, bmat, b1, b2, b3, W2h, W3h, (float*)d_out);
}
